// JunmaiLayer_5274219840157
// MI455X (gfx1250) — matrix operand order repaired and checked by reading, not compiled, not run on hardware
//
#include <hip/hip_runtime.h>
#include <hip/hip_bf16.h>
#include <math.h>

typedef _Float16 f16;
typedef __attribute__((ext_vector_type(16))) _Float16 v16h;
typedef __attribute__((ext_vector_type(8)))  _Float16 v8h;
typedef __attribute__((ext_vector_type(8)))  float    v8f;

#define NN      1536
#define NRBF    16
#define HID     64
#define IB      16      // i rows per block (one WMMA M tile)
#define JB      32      // j columns per outer step
#define EPS     1e-5f
#define CUT_UP  5.0f

// ---------------------------------------------------------------------------
// Kernel 1: transpose+convert K,Q (N, R, H) f32  ->  (N, H, R) f16
// so that a WMMA B-fragment lane load is 16 contiguous f16 (32 bytes).
// ---------------------------------------------------------------------------
__global__ void junmai_cvt_kernel(const float* __restrict__ K,
                                  const float* __restrict__ Q,
                                  f16* __restrict__ Kh,
                                  f16* __restrict__ Qh) {
    int idx = blockIdx.x * blockDim.x + threadIdx.x;      // idx = (j*64 + c)*16 + a
    if (idx >= NN * HID * NRBF) return;
    int a = idx & 15;
    int c = (idx >> 4) & 63;
    int j = idx >> 10;
    int src = (j * NRBF + a) * HID + c;
    Kh[idx] = (f16)K[src];
    Qh[idx] = (f16)Q[src];
}

// ---------------------------------------------------------------------------
// Kernel 2: fused basis construction + double GEMM (K and Q) + attention dot
//           + SiLU + 64->1 projection.
// Grid: NN/IB = 96 blocks of 256 threads (8 wave32s).
// Wave w: matrix m = w>>2 (0=K, 1=Q), hidden tile h = w&3 (cols h*16..h*16+15).
// ---------------------------------------------------------------------------
__global__ __launch_bounds__(256)
void junmai_main_kernel(const float* __restrict__ x,
                        const f16*  __restrict__ Kh,
                        const f16*  __restrict__ Qh,
                        const float* __restrict__ means,
                        const float* __restrict__ betas,
                        const float* __restrict__ w_out,
                        const float* __restrict__ b_out,
                        float* __restrict__ out) {
    // LDS: basis tiles A_b[b][i][j][a] as f16 (a contiguous), 3*16*32*16 = 48KB.
    // Reused after the main loop as the f32 accumulator exchange buffer (24KB).
    __shared__ __align__(32) unsigned char smem[3 * IB * JB * NRBF * sizeof(f16)];
    f16*   Abuf   = (f16*)smem;             // halves; b stride = 8192 halves
    float* accbuf = (float*)smem;           // [m][b][row16][col64]

    const int tid   = threadIdx.x;
    const int wave  = tid >> 5;
    const int lane  = tid & 31;
    const int m     = wave >> 2;            // 0 = K, 1 = Q
    const int h     = wave & 3;             // hidden 16-tile
    const int c0col = h * 16;
    const int lrow  = lane & 15;            // WMMA row / column within tile
    const int lhalf = lane >> 4;            // lane half selects j parity
    const int iblk  = blockIdx.x;

    const f16* __restrict__ Bmat = m ? Qh : Kh;

    v8f acc[3];
    #pragma unroll
    for (int b = 0; b < 3; ++b) acc[b] = (v8f){};

    const float invCut = 3.14159265358979f / CUT_UP;

    for (int jb = 0; jb < NN / JB; ++jb) {
        __syncthreads();   // previous step's LDS reads complete before refill

        // ---- cooperative basis tile build: 16i x 32j pairs, 2 per thread ----
        for (int p = tid; p < IB * JB; p += 256) {
            int iloc = p >> 5;
            int jloc = p & 31;
            int ig = iblk * IB + iloc;
            int jg = jb * JB + jloc;
            float dx = x[ig * 3 + 0] - x[jg * 3 + 0];
            float dy = x[ig * 3 + 1] - x[jg * 3 + 1];
            float dz = x[ig * 3 + 2] - x[jg * 3 + 2];
            float r2  = dx * dx + dy * dy + dz * dz + EPS;
            float r   = sqrtf(r2);
            float inv = 1.0f / r2;
            float d0 = dx * inv, d1 = dy * inv, d2 = dz * inv;
            float cut = (r < CUT_UP) ? 0.5f * (__cosf(r * invCut) + 1.0f) : 0.0f;
            float en  = __expf(-r);

            f16 vals[3][NRBF];
            #pragma unroll
            for (int a = 0; a < NRBF; ++a) {
                float t = en - means[a];
                float s = cut * __expf(-betas[a] * t * t);
                vals[0][a] = (f16)(s * d0);
                vals[1][a] = (f16)(s * d1);
                vals[2][a] = (f16)(s * d2);
            }
            int base = (iloc * JB + jloc) * NRBF;
            #pragma unroll
            for (int b = 0; b < 3; ++b) {
                v8h lo, hi;
                #pragma unroll
                for (int e = 0; e < 8; ++e) { lo[e] = vals[b][e]; hi[e] = vals[b][8 + e]; }
                *(v8h*)&Abuf[b * 8192 + base]     = lo;   // ds_store_b128
                *(v8h*)&Abuf[b * 8192 + base + 8] = hi;   // ds_store_b128
            }
        }
        __syncthreads();

        // ---- WMMA sweep: 16 k-steps, each consumes 2 j-columns (K=32) ----
        #pragma unroll 4
        for (int t = 0; t < 16; ++t) {
            int j0 = jb * JB + 2 * t;
            // B fragment: lane half picks j0/j0+1; 16 contiguous a-values (32B)
            const v16h bfrag =
                *(const v16h*)(Bmat + ((size_t)(j0 + lhalf) * HID + c0col + lrow) * NRBF);
            // A fragments from LDS: lo = (j=2t+lhalf, a=0..7), hi = (j=2t+lhalf, a=8..15)
            int abase = (lrow * JB + 2 * t + lhalf) * NRBF;
            #pragma unroll
            for (int b = 0; b < 3; ++b) {
                v8h lo = *(const v8h*)&Abuf[b * 8192 + abase];          // ds_load_b128
                v8h hi = *(const v8h*)&Abuf[b * 8192 + abase + 8];   // ds_load_b128
                v16h af = __builtin_shufflevector(lo, hi,
                    0, 1, 2, 3, 4, 5, 6, 7, 8, 9, 10, 11, 12, 13, 14, 15);
                acc[b] = __builtin_amdgcn_wmma_f32_16x16x32_f16(
                    false, af, false, bfrag, (short)0, acc[b], false, false);
            }
        }
    }

    __syncthreads();   // all WMMA A reads done; reuse LDS for f32 accumulators

    // ---- exchange accumulators: accbuf[(m*3+b)*16 + row][64] ----
    #pragma unroll
    for (int b = 0; b < 3; ++b) {
        #pragma unroll
        for (int v = 0; v < 8; ++v) {
            int row = v + (lhalf ? 8 : 0);
            accbuf[((m * 3 + b) * IB + row) * HID + c0col + lrow] = acc[b][v];
        }
    }
    __syncthreads();

    // ---- att = sum_b bk*bq ; out = SiLU(att) @ w_out.T + b_out ----
    if (tid < IB) {
        float sum = 0.0f;
        for (int c = 0; c < HID; ++c) {
            float att = 0.0f;
            #pragma unroll
            for (int b = 0; b < 3; ++b) {
                float bk = accbuf[((0 * 3 + b) * IB + tid) * HID + c];
                float bq = accbuf[((1 * 3 + b) * IB + tid) * HID + c];
                att += bk * bq;
            }
            float sig = 1.0f / (1.0f + __expf(-att));
            sum += att * sig * w_out[c];
        }
        out[iblk * IB + tid] = sum + b_out[0];
    }
}

// ---------------------------------------------------------------------------
extern "C" void kernel_launch(void* const* d_in, const int* in_sizes, int n_in,
                              void* d_out, int out_size, void* d_ws, size_t ws_size,
                              hipStream_t stream) {
    const float* x     = (const float*)d_in[0];
    const float* K     = (const float*)d_in[1];
    const float* Q     = (const float*)d_in[2];
    const float* means = (const float*)d_in[3];
    const float* betas = (const float*)d_in[4];
    const float* w_out = (const float*)d_in[5];
    const float* b_out = (const float*)d_in[6];
    float* out = (float*)d_out;

    f16* Kh = (f16*)d_ws;                                 // 3.1 MB
    f16* Qh = Kh + (size_t)NN * HID * NRBF;               // 3.1 MB

    int total = NN * HID * NRBF;
    junmai_cvt_kernel<<<(total + 255) / 256, 256, 0, stream>>>(K, Q, Kh, Qh);
    junmai_main_kernel<<<NN / IB, 256, 0, stream>>>(x, Kh, Qh, means, betas,
                                                    w_out, b_out, out);
}
